// Encoder_51230369907070
// MI455X (gfx1250) — matrix operand order repaired and checked by reading, not compiled, not run on hardware
//
#include <hip/hip_runtime.h>

// ---------------------------------------------------------------------------
// Problem constants (from the reference)
// ---------------------------------------------------------------------------
#define BATCH  4
#define SEQ    2048
#define DMODEL 1024
#define HEADS  16
#define DHEAD  64
#define MROWS  (BATCH * SEQ)        // 8192 token rows
#define QBLK   (SEQ / 16)           // 128 query blocks per (b,h)
#define SCALE  0.125f               // 1/sqrt(64)

typedef __attribute__((ext_vector_type(16))) __bf16          v16bf;
typedef __attribute__((ext_vector_type(8)))  float           v8f;
typedef __attribute__((ext_vector_type(8)))  unsigned short  ushort8;

union Frag {
  v16bf   bf;
  ushort8 h[2];
};

__device__ __forceinline__ unsigned short f32_to_bf16(float f) {
  unsigned int u = __builtin_bit_cast(unsigned int, f);
  u += 0x7FFFu + ((u >> 16) & 1u);   // round-to-nearest-even
  return (unsigned short)(u >> 16);
}

// ---------------------------------------------------------------------------
// Elementwise prologue kernels (cast / transpose into bf16 workspace)
// ---------------------------------------------------------------------------
__global__ __launch_bounds__(256) void cast_bf16_kernel(
    const float* __restrict__ in, unsigned short* __restrict__ out, long n) {
  long i = (long)blockIdx.x * 256 + threadIdx.x;
  if (i < n) out[i] = f32_to_bf16(in[i]);
}

// Build W_qkv^T : [3*DMODEL][DMODEL], row n = output feature, col k = input.
__global__ __launch_bounds__(256) void build_wqkv_t_kernel(
    const float* __restrict__ wq, const float* __restrict__ wk,
    const float* __restrict__ wv, unsigned short* __restrict__ out) {
  long i = (long)blockIdx.x * 256 + threadIdx.x;   // i = n*1024 + k
  int n = (int)(i >> 10), k = (int)(i & 1023);
  int sel = n >> 10;
  int nn  = n & 1023;
  int hh  = nn >> 6, e = nn & 63;
  const float* w = (sel == 0) ? wq : ((sel == 1) ? wk : wv);
  out[i] = f32_to_bf16(w[((size_t)hh * DMODEL + k) * DHEAD + e]);
}

// Generic [K][N] -> [N][K] transpose with fp32->bf16 cast
__global__ __launch_bounds__(256) void transpose_cast_kernel(
    const float* __restrict__ in, unsigned short* __restrict__ out, int K, int N) {
  long i = (long)blockIdx.x * 256 + threadIdx.x;   // i over N*K, [n][k]
  int n = (int)(i / K), k = (int)(i % K);
  out[i] = f32_to_bf16(in[(size_t)k * N + n]);
}

// V^T : qkv[B*S][3072] (V at col 2048 + h*64 + e) -> vt[B][H][dh][S]
__global__ __launch_bounds__(256) void build_vt_kernel(
    const unsigned short* __restrict__ qkv, unsigned short* __restrict__ vt) {
  long i = (long)blockIdx.x * 256 + threadIdx.x;   // i = ((b*H+h)*64+e)*S + s
  int  s  = (int)(i & (SEQ - 1));
  long r  = i >> 11;
  int  e  = (int)(r & 63);
  long bh = r >> 6;
  int  h  = (int)(bh & (HEADS - 1));
  int  b  = (int)(bh >> 4);
  vt[i] = qkv[(size_t)(b * SEQ + s) * (3 * DMODEL) + 2 * DMODEL + h * DHEAD + e];
}

// ---------------------------------------------------------------------------
// bf16 WMMA GEMM:  C[M,N] = epilogue( A[M,K] * Bt[N,K]^T )
// Block = 256 thr (8 waves), tile 128x256x32; wave tile 64x64 (4x4 WMMA tiles)
// -> 16 WMMA per k-step per wave, 1:1 ds_load_b128 : wmma ratio.
// Copy phase fully unrolled & strength-reduced: no loops / 64-bit muls in the
// k-loop, just clause'd global_load_b128 -> ds_store_b128.
// ---------------------------------------------------------------------------
#define BM 128
#define BN 256
#define BK 32
#define LDSK 40   // padded row stride (elems): 80B rows -> 16B aligned frags

__global__ __launch_bounds__(256) void gemm_bf16_kernel(
    const unsigned short* __restrict__ A,    // [M][K] bf16
    const unsigned short* __restrict__ Bt,   // [N][K] bf16 (B transposed)
    int M, int N, int K,
    const float* __restrict__ bias,          // [N] or nullptr
    const float* __restrict__ resid,         // [M][N] f32 or nullptr
    float* __restrict__ Cf,                  // f32 out or nullptr
    unsigned short* __restrict__ Cb,         // bf16 out or nullptr
    int relu) {
  __shared__ unsigned short ldsA[BM * LDSK];
  __shared__ unsigned short ldsB[BN * LDSK];

  const int tid  = threadIdx.x;
  const int wave = tid >> 5;
  const int lane = tid & 31;
  const int half = lane >> 4;
  const int l16  = lane & 15;

  const int bm = blockIdx.x * BM;
  const int bn = blockIdx.y * BN;
  const int wm = (wave >> 2) * 64;   // 2 waves along M
  const int wn = (wave & 3) * 64;    // 4 waves along N

  v8f acc[4][4] = {};

  // --- copy-phase addressing, computed once (k-loop just adds BK) ---
  const int crow = tid >> 1;          // 0..127
  const int coff = (tid & 1) * 16;    // 0 or 16 (elems)
  const unsigned short* ga  = A  + (size_t)(bm + crow) * K + coff;
  const unsigned short* gb0 = Bt + (size_t)(bn + crow) * K + coff;
  const unsigned short* gb1 = gb0 + (size_t)128 * K;
  unsigned short* la  = &ldsA[crow * LDSK + coff];
  unsigned short* lb0 = &ldsB[crow * LDSK + coff];
  unsigned short* lb1 = lb0 + 128 * LDSK;

  // --- fragment read addressing (invariant) ---
  const unsigned short* fa = &ldsA[(wm + l16) * LDSK + half * 8];
  const unsigned short* fb = &ldsB[(wn + l16) * LDSK + half * 8];

  for (int k0 = 0; k0 < K; k0 += BK) {
    // ---- global -> LDS: A 128x32 (32B/thr), B 256x32 (64B/thr) ----
    *(ushort8*)la        = *(const ushort8*)ga;
    *(ushort8*)(la + 8)  = *(const ushort8*)(ga + 8);
    *(ushort8*)lb0       = *(const ushort8*)gb0;
    *(ushort8*)(lb0 + 8) = *(const ushort8*)(gb0 + 8);
    *(ushort8*)lb1       = *(const ushort8*)gb1;
    *(ushort8*)(lb1 + 8) = *(const ushort8*)(gb1 + 8);
    ga += BK; gb0 += BK; gb1 += BK;
    __syncthreads();

    // ---- fragments + 16 WMMA ----
    Frag af[4];
#pragma unroll
    for (int i = 0; i < 4; ++i) {
      const unsigned short* p = fa + i * 16 * LDSK;
      af[i].h[0] = *(const ushort8*)p;          // K = half*8 .. +7
      af[i].h[1] = *(const ushort8*)(p + 16);   // K = 16+half*8 .. +7
    }
#pragma unroll
    for (int j = 0; j < 4; ++j) {
      Frag bf_;
      const unsigned short* p = fb + j * 16 * LDSK;
      bf_.h[0] = *(const ushort8*)p;            // K = half*8 .. +7
      bf_.h[1] = *(const ushort8*)(p + 16);     // K = 16+half*8 .. +7
#pragma unroll
      for (int i = 0; i < 4; ++i)
        acc[i][j] = __builtin_amdgcn_wmma_f32_16x16x32_bf16(
            false, af[i].bf, false, bf_.bf, (short)0, acc[i][j], false, false);
    }
    __syncthreads();
  }

  // ---- epilogue: bias / residual / relu / store ----
  for (int i = 0; i < 4; ++i) {
    for (int j = 0; j < 4; ++j) {
      const int col = bn + wn + j * 16 + l16;
      const float bv = bias ? bias[col] : 0.f;
      for (int r = 0; r < 8; ++r) {
        const int rrow = bm + wm + i * 16 + half * 8 + r;
        float v = acc[i][j][r] + bv;
        if (resid) v += resid[(size_t)rrow * N + col];
        if (relu)  v = v > 0.f ? v : 0.f;
        if (Cf) Cf[(size_t)rrow * N + col] = v;
        if (Cb) Cb[(size_t)rrow * N + col] = f32_to_bf16(v);
      }
    }
  }
}

// ---------------------------------------------------------------------------
// Flash attention (causal). One wave per (b, h, 16-row query block).
// qkv: [B*S][3072] bf16 (Q|K|V), vt: [B][H][dh][S] bf16, out: [B*S][1024] bf16
// ---------------------------------------------------------------------------
#define PSTR 40

__global__ __launch_bounds__(256) void attn_kernel(
    const unsigned short* __restrict__ qkv,
    const unsigned short* __restrict__ vt,
    unsigned short* __restrict__ out) {
  __shared__ unsigned short ldsP[8][16 * PSTR];

  const int tid  = threadIdx.x;
  const int wave = tid >> 5;
  const int lane = tid & 31;
  const int half = lane >> 4;
  const int l16  = lane & 15;

  const int task = blockIdx.x * 8 + wave;       // B*H*QBLK tasks total
  const int b    = task >> 11;                  // / (HEADS*QBLK)
  const int rem  = task & 2047;
  const int h    = rem >> 7;
  const int qb   = rem & (QBLK - 1);

  const int    brow = b * SEQ;
  const int    hcol = h * DHEAD;
  const size_t ldq  = 3 * DMODEL;

  // Q fragments for dh chunks kk=0,1 (A layout, held for the whole pass)
  Frag qf[2];
  {
    const unsigned short* qp =
        qkv + (size_t)(brow + qb * 16 + l16) * ldq + hcol;
#pragma unroll
    for (int kk = 0; kk < 2; ++kk) {
      const unsigned short* p = qp + kk * 32 + half * 8;
      qf[kk].h[0] = *(const ushort8*)p;
      qf[kk].h[1] = *(const ushort8*)(p + 16);
    }
  }

  v8f   o[4] = {};
  float m_r[8], l_r[8];
#pragma unroll
  for (int r = 0; r < 8; ++r) { m_r[r] = -1e30f; l_r[r] = 0.f; }

  const unsigned short* vtp = vt + (size_t)((b * HEADS + h) * DHEAD) * SEQ;
  const int kend = qb * 16 + 16;   // exclusive key bound (causal)

  for (int t0 = 0; t0 < kend; t0 += 32) {
    // ---- S = Q K^T for two 16-key tiles ----
    v8f st[2];
#pragma unroll
    for (int kt = 0; kt < 2; ++kt) {
      const int kb = t0 + kt * 16;
      v8f c = {};
#pragma unroll
      for (int kk = 0; kk < 2; ++kk) {
        Frag kf;
        const unsigned short* kp =
            qkv + (size_t)(brow + kb + l16) * ldq + DMODEL + hcol +
            kk * 32 + half * 8;
        kf.h[0] = *(const ushort8*)kp;
        kf.h[1] = *(const ushort8*)(kp + 16);
        c = __builtin_amdgcn_wmma_f32_16x16x32_bf16(
            false, qf[kk].bf, false, kf.bf, (short)0, c, false, false);
      }
      st[kt] = c;
    }

    // ---- online softmax (per row r; rows r+half*8) ----
    float alpha[8];
#pragma unroll
    for (int r = 0; r < 8; ++r) {
      const int qi = qb * 16 + r + half * 8;
      float a0 = st[0][r] * SCALE;
      float a1 = st[1][r] * SCALE;
      if (t0 + l16 > qi)      a0 = -1e30f;
      if (t0 + 16 + l16 > qi) a1 = -1e30f;
      float mx = fmaxf(a0, a1);
      for (int off = 1; off < 16; off <<= 1)
        mx = fmaxf(mx, __shfl_xor(mx, off, 16));
      const float nm = fmaxf(m_r[r], mx);
      const float al = __expf(m_r[r] - nm);
      const float p0 = __expf(a0 - nm);
      const float p1 = __expf(a1 - nm);
      float rs = p0 + p1;
      for (int off = 1; off < 16; off <<= 1)
        rs += __shfl_xor(rs, off, 16);
      l_r[r]   = l_r[r] * al + rs;
      m_r[r]   = nm;
      alpha[r] = al;
      unsigned short* pr = &ldsP[wave][(r + half * 8) * PSTR];
      pr[l16]      = f32_to_bf16(p0);
      pr[16 + l16] = f32_to_bf16(p1);
    }
#pragma unroll
    for (int t = 0; t < 4; ++t)
#pragma unroll
      for (int r = 0; r < 8; ++r) o[t][r] *= alpha[r];

    // P: C layout -> A layout via per-wave LDS tile (same wave, in order)
    asm volatile("" ::: "memory");
    Frag pf;
    {
      const unsigned short* pp = &ldsP[wave][l16 * PSTR + half * 8];
      pf.h[0] = *(const ushort8*)pp;
      pf.h[1] = *(const ushort8*)(pp + 16);
    }

    // ---- O += P * V (4 dh tiles), V^T rows are contiguous in vt ----
#pragma unroll
    for (int n0 = 0; n0 < 4; ++n0) {
      Frag vf;
      const unsigned short* vp =
          vtp + (size_t)(n0 * 16 + l16) * SEQ + t0 + half * 8;
      vf.h[0] = *(const ushort8*)vp;
      vf.h[1] = *(const ushort8*)(vp + 16);
      o[n0] = __builtin_amdgcn_wmma_f32_16x16x32_bf16(
          false, pf.bf, false, vf.bf, (short)0, o[n0], false, false);
    }
  }

  // ---- normalize and store (concat-head layout [B*S][H*dh]) ----
  float invl[8];
#pragma unroll
  for (int r = 0; r < 8; ++r) invl[r] = 1.0f / l_r[r];
  for (int n0 = 0; n0 < 4; ++n0) {
    for (int r = 0; r < 8; ++r) {
      const int rrow = brow + qb * 16 + r + half * 8;
      out[(size_t)rrow * DMODEL + hcol + n0 * 16 + l16] =
          f32_to_bf16(o[n0][r] * invl[r]);
    }
  }
}

// ---------------------------------------------------------------------------
// LayerNorm: one block per row (D=1024, 4 elems/thread). Emits f32 (+ bf16).
// ---------------------------------------------------------------------------
__global__ __launch_bounds__(256) void layernorm_kernel(
    const float* __restrict__ x, const float* __restrict__ g,
    const float* __restrict__ be, float* __restrict__ yf,
    unsigned short* __restrict__ yb) {
  __shared__ float rs[8], rs2[8];
  const int row = blockIdx.x;
  const int tid = threadIdx.x;
  const float* xr = x + (size_t)row * DMODEL;

  float v[4], s = 0.f, s2 = 0.f;
#pragma unroll
  for (int i = 0; i < 4; ++i) {
    v[i] = xr[tid + i * 256];
    s += v[i]; s2 += v[i] * v[i];
  }
  for (int off = 16; off > 0; off >>= 1) {
    s  += __shfl_xor(s, off, 32);
    s2 += __shfl_xor(s2, off, 32);
  }
  if ((tid & 31) == 0) { rs[tid >> 5] = s; rs2[tid >> 5] = s2; }
  __syncthreads();
  s = 0.f; s2 = 0.f;
#pragma unroll
  for (int w = 0; w < 8; ++w) { s += rs[w]; s2 += rs2[w]; }

  const float mean = s * (1.0f / DMODEL);
  const float var  = s2 * (1.0f / DMODEL) - mean * mean;
  const float inv  = rsqrtf(var + 1e-6f);
#pragma unroll
  for (int i = 0; i < 4; ++i) {
    const int c = tid + i * 256;
    const float ov = (v[i] - mean) * inv * g[c] + be[c];
    if (yf) yf[(size_t)row * DMODEL + c] = ov;
    if (yb) yb[(size_t)row * DMODEL + c] = f32_to_bf16(ov);
  }
}

// ---------------------------------------------------------------------------
// Host orchestration
// ---------------------------------------------------------------------------
extern "C" void kernel_launch(void* const* d_in, const int* in_sizes, int n_in,
                              void* d_out, int out_size, void* d_ws, size_t ws_size,
                              hipStream_t stream) {
  const float* emb    = (const float*)d_in[0];
  const float* W_Q    = (const float*)d_in[1];
  const float* W_K    = (const float*)d_in[2];
  const float* W_V    = (const float*)d_in[3];
  const float* proj_w = (const float*)d_in[4];
  const float* proj_b = (const float*)d_in[5];
  const float* ln1_g  = (const float*)d_in[6];
  const float* ln1_b  = (const float*)d_in[7];
  const float* ff_w1  = (const float*)d_in[8];
  const float* ff_b1  = (const float*)d_in[9];
  const float* ff_w2  = (const float*)d_in[10];
  const float* ff_b2  = (const float*)d_in[11];
  const float* ln2_g  = (const float*)d_in[12];
  const float* ln2_b  = (const float*)d_in[13];

  char* ws = (char*)d_ws;
  size_t off = 0;
  auto alloc = [&](size_t bytes) -> void* {
    void* p = ws + off;
    off += (bytes + 255) & ~(size_t)255;
    return p;
  };
  unsigned short* x_bf  = (unsigned short*)alloc((size_t)MROWS * DMODEL * 2);
  unsigned short* wqkv  = (unsigned short*)alloc((size_t)3 * DMODEL * DMODEL * 2);
  unsigned short* wproj = (unsigned short*)alloc((size_t)DMODEL * DMODEL * 2);
  unsigned short* wff1  = (unsigned short*)alloc((size_t)4 * DMODEL * DMODEL * 2);
  unsigned short* wff2  = (unsigned short*)alloc((size_t)4 * DMODEL * DMODEL * 2);
  unsigned short* qkv   = (unsigned short*)alloc((size_t)MROWS * 3 * DMODEL * 2);
  unsigned short* vt    = (unsigned short*)alloc((size_t)MROWS * DMODEL * 2);
  unsigned short* attn  = (unsigned short*)alloc((size_t)MROWS * DMODEL * 2);
  float*          y1    = (float*)alloc((size_t)MROWS * DMODEL * 4);
  float*          x1f   = (float*)alloc((size_t)MROWS * DMODEL * 4);
  unsigned short* x1b   = (unsigned short*)alloc((size_t)MROWS * DMODEL * 2);
  unsigned short* h1    = (unsigned short*)alloc((size_t)MROWS * 4 * DMODEL * 2);
  float*          y2    = (float*)alloc((size_t)MROWS * DMODEL * 4);

  const long nx = (long)MROWS * DMODEL;   // 8388608

  // 1) casts / transposes
  cast_bf16_kernel<<<nx / 256, 256, 0, stream>>>(emb, x_bf, nx);
  build_wqkv_t_kernel<<<(3 * DMODEL * DMODEL) / 256, 256, 0, stream>>>(W_Q, W_K, W_V, wqkv);
  transpose_cast_kernel<<<(DMODEL * DMODEL) / 256, 256, 0, stream>>>(proj_w, wproj, DMODEL, DMODEL);
  transpose_cast_kernel<<<(4 * DMODEL * DMODEL) / 256, 256, 0, stream>>>(ff_w1, wff1, DMODEL, 4 * DMODEL);
  transpose_cast_kernel<<<(4 * DMODEL * DMODEL) / 256, 256, 0, stream>>>(ff_w2, wff2, 4 * DMODEL, DMODEL);

  // 2) fused QKV projection: [8192,1024] x [1024,3072] -> [8192,3072] bf16
  gemm_bf16_kernel<<<dim3(MROWS / BM, (3 * DMODEL) / BN), 256, 0, stream>>>(
      x_bf, wqkv, MROWS, 3 * DMODEL, DMODEL, nullptr, nullptr, nullptr, qkv, 0);

  // 3) V^T for the PV matmul
  build_vt_kernel<<<nx / 256, 256, 0, stream>>>(qkv, vt);

  // 4) causal flash attention
  attn_kernel<<<(BATCH * HEADS * QBLK) / 8, 256, 0, stream>>>(qkv, vt, attn);

  // 5) output projection + bias + residual(embeddings) -> y1 (f32)
  gemm_bf16_kernel<<<dim3(MROWS / BM, DMODEL / BN), 256, 0, stream>>>(
      attn, wproj, MROWS, DMODEL, DMODEL, proj_b, emb, y1, nullptr, 0);

  // 6) LN1 -> x1 (f32 + bf16)
  layernorm_kernel<<<MROWS, 256, 0, stream>>>(y1, ln1_g, ln1_b, x1f, x1b);

  // 7) FFN up + ReLU -> h1 (bf16)
  gemm_bf16_kernel<<<dim3(MROWS / BM, (4 * DMODEL) / BN), 256, 0, stream>>>(
      x1b, wff1, MROWS, 4 * DMODEL, DMODEL, ff_b1, nullptr, nullptr, h1, 1);

  // 8) FFN down + bias + residual(x1) -> y2 (f32)
  gemm_bf16_kernel<<<dim3(MROWS / BM, DMODEL / BN), 256, 0, stream>>>(
      h1, wff2, MROWS, DMODEL, 4 * DMODEL, ff_b2, x1f, y2, nullptr, 0);

  // 9) LN2 -> output (f32)
  layernorm_kernel<<<MROWS, 256, 0, stream>>>(y2, ln2_g, ln2_b, (float*)d_out, nullptr);
}
